// ModelGraphCoordinationNet_66614942761538
// MI455X (gfx1250) — matrix operand order repaired and checked by reading, not compiled, not run on hardware
//
#include <hip/hip_runtime.h>

// ---------------------------------------------------------------------------
// CDNA5 (gfx1250) implementation of the coordination-net GNN forward pass.
// Edge CGConv matmuls are done with v_wmma_f32_16x16x32_f16: 16 edges/wave as
// the A matrix, weights pre-packed into WMMA-B fragment layout (f16), f32
// accumulation, gated message epilogue with global f32 atomics for segment_sum.
// ---------------------------------------------------------------------------

typedef __attribute__((ext_vector_type(16))) _Float16 v16h;
typedef __attribute__((ext_vector_type(8)))  float    v8f;
typedef _Float16 half_t;

#define CDIV(a, b) (((a) + (b) - 1) / (b))

namespace {
constexpr int kNSite = 50000, kNCe = 50000, kNLig = 100000;
constexpr int kESS = 200000, kELC = 200000, kECL = 200000, kECS = 100000;
constexpr int kG = 2000;
constexpr int kDSite = 80, kDCe = 112, kDLig = 112;
}

union AFrag { v16h v; _Float16 h[16]; };
union CFrag { v8f  v; float    f[8];  };

__device__ __forceinline__ float dev_elu(float x) {
  return x > 0.f ? x : expm1f(x);
}
__device__ __forceinline__ float dev_sigmoid(float x) {
  return 1.f / (1.f + expf(-x));
}
__device__ __forceinline__ float dev_softplus(float x) {
  return x > 15.f ? x : log1pf(expf(x));
}
// Gaussian RBF over 16 centers in [0,1], then linear (W:[16,16], b:[16]).
__device__ __forceinline__ float dev_rbf(float x, int col,
                                         const float* __restrict__ W,
                                         const float* __restrict__ b) {
  float acc = b[col];
#pragma unroll
  for (int k = 0; k < 16; ++k) {
    float d = (x - (float)k * (1.f / 15.f)) * 15.f;
    acc += expf(-d * d) * W[k * 16 + col];
  }
  return acc;
}

// ---------------------------- feature builders -----------------------------

__global__ void build_site(const int* __restrict__ el, const int* __restrict__ ox,
                           const float* __restrict__ embE, const float* __restrict__ embO,
                           float* __restrict__ x) {
  int t = blockIdx.x * blockDim.x + threadIdx.x;
  if (t >= kNSite * kDSite) return;
  int n = t / kDSite, c = t % kDSite;
  x[t] = (c < 64) ? embE[el[n] * 64 + c] : embO[ox[n] * 16 + (c - 64)];
}

__global__ void build_ce(const int* __restrict__ el, const int* __restrict__ ox,
                         const int* __restrict__ geo, const float* __restrict__ csm,
                         const float* __restrict__ embE, const float* __restrict__ embO,
                         const float* __restrict__ embG,
                         const float* __restrict__ Wc, const float* __restrict__ bc,
                         float* __restrict__ x) {
  int t = blockIdx.x * blockDim.x + threadIdx.x;
  if (t >= kNCe * kDCe) return;
  int n = t / kDCe, c = t % kDCe;
  float v;
  if (c < 64)       v = embE[el[n] * 64 + c];
  else if (c < 80)  v = embO[ox[n] * 16 + (c - 64)];
  else if (c < 96)  v = embG[geo[n] * 16 + (c - 80)];
  else              v = dev_rbf(csm[n], c - 96, Wc, bc);
  x[t] = v;
}

__global__ void build_lig(const int* __restrict__ el, const int* __restrict__ ox,
                          const float* __restrict__ dist, const float* __restrict__ ang,
                          const float* __restrict__ embE, const float* __restrict__ embO,
                          const float* __restrict__ Wd, const float* __restrict__ bd,
                          const float* __restrict__ Wa, const float* __restrict__ ba,
                          float* __restrict__ x) {
  int t = blockIdx.x * blockDim.x + threadIdx.x;
  if (t >= kNLig * kDLig) return;
  int n = t / kDLig, c = t % kDLig;
  float v;
  if (c < 64)       v = embE[el[n] * 64 + c];
  else if (c < 80)  v = embO[ox[n] * 16 + (c - 64)];
  else if (c < 96)  v = dev_rbf(dist[n], c - 80, Wd, bd);
  else              v = dev_rbf(ang[n],  c - 96, Wa, ba);
  x[t] = v;
}

__global__ void edge_rbf(const float* __restrict__ ea, float* __restrict__ out, int E,
                         const float* __restrict__ W, const float* __restrict__ b) {
  int t = blockIdx.x * blockDim.x + threadIdx.x;
  if (t >= E * 16) return;
  out[t] = dev_rbf(ea[t / 16], t % 16, W, b);
}

// ----------------------- WMMA-B fragment weight packing --------------------
// B packing used here (32x16 tile, contiguous K per lane half, same order as the A loads): lane holds column N = lane&15,
// half j of v16h holds K = kt*32 + (lane>=16 ? 16 : 0) + j.

__global__ void pack_b(const float* __restrict__ W, half_t* __restrict__ out,
                       int Zr, int D, int KT, int NT) {
  int t = blockIdx.x * blockDim.x + threadIdx.x;
  if (t >= NT * KT * 512) return;
  int j = t & 15, lane = (t >> 4) & 31;
  int kt = (t >> 9) % KT, nt = (t >> 9) / KT;
  int n = nt * 16 + (lane & 15);
  int k = kt * 32 + ((lane >= 16) ? 16 : 0) + j;
  float v = (k < Zr && n < D) ? W[k * D + n] : 0.f;
  out[t] = (half_t)v;
}

// Pack [W_rel(112x80) ; W_root(80x80)] -> single 192x80 B-fragment buffer.
__global__ void pack_b_gc(const float* __restrict__ Wrel, const float* __restrict__ Wroot,
                          half_t* __restrict__ out) {
  const int KT = 6, NT = 5;
  int t = blockIdx.x * blockDim.x + threadIdx.x;
  if (t >= NT * KT * 512) return;
  int j = t & 15, lane = (t >> 4) & 31;
  int kt = (t >> 9) % KT;
  int n = ((t >> 9) / KT) * 16 + (lane & 15);
  int k = kt * 32 + ((lane >= 16) ? 16 : 0) + j;
  float v = 0.f;
  if (k < 112)      v = Wrel[k * 80 + n];
  else if (k < 192) v = Wroot[(k - 112) * 80 + n];
  out[t] = (half_t)v;
}

// ------------------------------ CGConv edges -------------------------------
// One wave per 16-edge tile. z = [x_dst[dst] (DD) | x_src[src] (DS) | ea (16)]
// zero-padded to KT*32. Output dim = DD = NT*16 exactly.

template <int DD, int DS, int HASE, int KT, int NT>
__global__ __launch_bounds__(128) void cgconv_edge(
    const int* __restrict__ ei, int E,
    const float* __restrict__ xdst, const float* __restrict__ xsrc,
    const float* __restrict__ eat,
    const half_t* __restrict__ Bf, const half_t* __restrict__ Bs,
    const float* __restrict__ bf, const float* __restrict__ bs,
    float* __restrict__ agg) {
  const int lane = threadIdx.x & 31;
  const int wave = threadIdx.x >> 5;
  const long long base_e = ((long long)blockIdx.x * 4 + wave) * 16;
  if (base_e >= E) return;  // wave-uniform: EXEC stays all-ones for WMMA

  const int m  = lane & 15;
  const int hi = lane >> 4;
  int e = (int)base_e + m;
  if (e > E - 1) e = E - 1;                 // clamp (epilogue is guarded)
  const int src = ei[e];
  const int dst = ei[E + e];

  const v8f zero = {0.f, 0.f, 0.f, 0.f, 0.f, 0.f, 0.f, 0.f};
  CFrag accF[NT], accS[NT];
#pragma unroll
  for (int nt = 0; nt < NT; ++nt) { accF[nt].v = zero; accS[nt].v = zero; }

  const int kbase = hi * 16;  // contiguous K per lane half (matches B packing)
#pragma unroll
  for (int kt = 0; kt < KT; ++kt) {
    AFrag a;
#pragma unroll
    for (int j = 0; j < 16; ++j) {
      int k = kt * 32 + kbase + j;
      float v;
      if (k < DD)                         v = xdst[(long long)dst * DD + k];
      else if (k < DD + DS)               v = xsrc[(long long)src * DS + (k - DD)];
      else if (HASE && k < DD + DS + 16)  v = eat[(long long)e * 16 + (k - DD - DS)];
      else                                v = 0.f;
      a.h[j] = (_Float16)v;
    }
#pragma unroll
    for (int nt = 0; nt < NT; ++nt) {
      const long long fo = (((long long)nt * KT + kt) * 32 + lane) * 16;
      v16h bfr = *(const v16h*)(Bf + fo);
      v16h bsr = *(const v16h*)(Bs + fo);
      accF[nt].v = __builtin_amdgcn_wmma_f32_16x16x32_f16(
          false, a.v, false, bfr, (short)0, accF[nt].v, false, false);
      accS[nt].v = __builtin_amdgcn_wmma_f32_16x16x32_f16(
          false, a.v, false, bsr, (short)0, accS[nt].v, false, false);
    }
  }

  // C layout: vgpr i -> row M = hi*8 + i, col N = lane&15.
  int dstrow[8]; bool evalid[8];
#pragma unroll
  for (int i = 0; i < 8; ++i) {
    long long e2 = base_e + hi * 8 + i;
    evalid[i] = (e2 < E);
    dstrow[i] = ei[E + (evalid[i] ? (int)e2 : (E - 1))];
  }
#pragma unroll
  for (int nt = 0; nt < NT; ++nt) {
    const int col = nt * 16 + m;
    const float bfv = bf[col], bsv = bs[col];
#pragma unroll
    for (int i = 0; i < 8; ++i) {
      if (!evalid[i]) continue;
      float msg = dev_sigmoid(accF[nt].f[i] + bfv) * dev_softplus(accS[nt].f[i] + bsv);
      atomicAdd(&agg[(long long)dstrow[i] * DD + col], msg);
    }
  }
}

// residual + optional ELU, in place on the aggregation buffer
__global__ void residual_update(const float* __restrict__ xin, float* __restrict__ io,
                                int n, int act) {
  int t = blockIdx.x * blockDim.x + threadIdx.x;
  if (t >= n) return;
  float v = xin[t] + io[t];
  io[t] = act ? dev_elu(v) : v;
}

// --------------------------- final GraphConv -------------------------------

__global__ void scatter_cs(const int* __restrict__ ei, const float* __restrict__ xce,
                           float* __restrict__ aggcs) {
  int t = blockIdx.x * blockDim.x + threadIdx.x;
  if (t >= kECS * kDCe) return;
  int e = t / kDCe, c = t % kDCe;
  atomicAdd(&aggcs[(long long)ei[kECS + e] * kDCe + c],
            xce[(long long)ei[e] * kDCe + c]);
}

// h = elu([aggcs | x_site] @ [W_rel ; W_root]); z = 192, D = 80.
__global__ __launch_bounds__(128) void gc_node_gemm(
    const float* __restrict__ aggcs, const float* __restrict__ xsite,
    const half_t* __restrict__ Bp, float* __restrict__ h) {
  const int KT = 6, NT = 5;
  const int lane = threadIdx.x & 31;
  const int wave = threadIdx.x >> 5;
  const long long base_n = ((long long)blockIdx.x * 4 + wave) * 16;
  if (base_n >= kNSite) return;
  const int m = lane & 15, hi = lane >> 4;
  int node = (int)base_n + m;
  if (node > kNSite - 1) node = kNSite - 1;

  const v8f zero = {0.f, 0.f, 0.f, 0.f, 0.f, 0.f, 0.f, 0.f};
  CFrag acc[NT];
#pragma unroll
  for (int nt = 0; nt < NT; ++nt) acc[nt].v = zero;

  const int kbase = hi * 16;
#pragma unroll
  for (int kt = 0; kt < KT; ++kt) {
    AFrag a;
#pragma unroll
    for (int j = 0; j < 16; ++j) {
      int k = kt * 32 + kbase + j;
      float v = (k < 112) ? aggcs[(long long)node * 112 + k]
                          : xsite[(long long)node * 80 + (k - 112)];
      a.h[j] = (_Float16)v;
    }
#pragma unroll
    for (int nt = 0; nt < NT; ++nt) {
      v16h b = *(const v16h*)(Bp + (((long long)nt * KT + kt) * 32 + lane) * 16);
      acc[nt].v = __builtin_amdgcn_wmma_f32_16x16x32_f16(
          false, a.v, false, b, (short)0, acc[nt].v, false, false);
    }
  }
#pragma unroll
  for (int nt = 0; nt < NT; ++nt) {
    const int col = nt * 16 + m;
#pragma unroll
    for (int i = 0; i < 8; ++i) {
      long long n2 = base_n + hi * 8 + i;
      if (n2 < kNSite) h[n2 * 80 + col] = dev_elu(acc[nt].f[i]);
    }
  }
}

// ------------------------------- pooling -----------------------------------

__global__ void pool_sum(const int* __restrict__ batch, const float* __restrict__ h,
                         float* __restrict__ pooled) {
  int t = blockIdx.x * blockDim.x + threadIdx.x;
  if (t >= kNSite * 80) return;
  atomicAdd(&pooled[(long long)batch[t / 80] * 80 + (t % 80)], h[t]);
}
__global__ void pool_cnt(const int* __restrict__ batch, float* __restrict__ cnt) {
  int t = blockIdx.x * blockDim.x + threadIdx.x;
  if (t >= kNSite) return;
  atomicAdd(&cnt[batch[t]], 1.f);
}
__global__ void pool_div(float* __restrict__ pooled, const float* __restrict__ cnt) {
  int t = blockIdx.x * blockDim.x + threadIdx.x;
  if (t >= kG * 80) return;
  pooled[t] /= fmaxf(cnt[t / 80], 1.f);
}

// ------------------------------ dense head ---------------------------------

__global__ void dense_layer(const float* __restrict__ in, const float* __restrict__ W,
                            const float* __restrict__ b, float* __restrict__ out,
                            int N, int din, int dout, int act) {
  int t = blockIdx.x * blockDim.x + threadIdx.x;
  if (t >= N * dout) return;
  int n = t / dout, o = t % dout;
  float acc = b[o];
  for (int k = 0; k < din; ++k) acc += in[(long long)n * din + k] * W[(long long)k * dout + o];
  out[t] = act ? dev_elu(acc) : acc;
}

__global__ void scale_out(const float* __restrict__ in, const float* __restrict__ meanp,
                          const float* __restrict__ stdp, float* __restrict__ out, int n) {
  int t = blockIdx.x * blockDim.x + threadIdx.x;
  if (t >= n) return;
  out[t] = in[t] * stdp[0] + meanp[0];
}

// ------------------------------- launcher ----------------------------------

extern "C" void kernel_launch(void* const* d_in, const int* in_sizes, int n_in,
                              void* d_out, int out_size, void* d_ws, size_t ws_size,
                              hipStream_t stream) {
  (void)in_sizes; (void)n_in; (void)out_size; (void)ws_size;
  const int* site_el = (const int*)d_in[0];
  const int* site_ox = (const int*)d_in[1];
  const int* ce_el   = (const int*)d_in[2];
  const int* ce_ox   = (const int*)d_in[3];
  const int* ce_geo  = (const int*)d_in[4];
  const float* ce_csm = (const float*)d_in[5];
  const int* lig_el  = (const int*)d_in[6];
  const int* lig_ox  = (const int*)d_in[7];
  const float* lig_d = (const float*)d_in[8];
  const float* lig_a = (const float*)d_in[9];
  const int* ei_ss = (const int*)d_in[10];
  const int* ei_lc = (const int*)d_in[11];
  const int* ei_cl = (const int*)d_in[12];
  const int* ei_cs = (const int*)d_in[13];
  const float* ea_lc = (const float*)d_in[14];
  const float* ea_cl = (const float*)d_in[15];
  const int* batch   = (const int*)d_in[16];
  // params flattened in JAX pytree (sorted-dict) order after the 17 data inputs
  const int PB = 17;
  auto P = [&](int i) { return (const float*)d_in[PB + i]; };
  // per conv layer L: cl.{Wf,Ws,bf,bs}=L*12+0..3, lc=+4..7, ss=+8..11
  // dense (W,b)x4 = 24..31; emb_element=32, emb_geom=33, emb_ox=34
  // gc.W_rel=35, gc.W_root=36; rbf_ang{W,b}=37,38; rbf_csm=39,40;
  // rbf_d1=41,42; rbf_d2=43,44; scaler_mean=45, scaler_std=46

  char* ws = (char*)d_ws;
  size_t off = 0;
  auto alloc = [&](size_t bytes) -> void* {
    void* p = ws + off; off += (bytes + 255) & ~(size_t)255; return p;
  };
  float* xsA = (float*)alloc((size_t)kNSite * kDSite * 4);
  float* xsB = (float*)alloc((size_t)kNSite * kDSite * 4);
  float* xcA = (float*)alloc((size_t)kNCe * kDCe * 4);
  float* xcB = (float*)alloc((size_t)kNCe * kDCe * 4);
  float* xlA = (float*)alloc((size_t)kNLig * kDLig * 4);
  float* xlB = (float*)alloc((size_t)kNLig * kDLig * 4);
  float* elc = (float*)alloc((size_t)kELC * 16 * 4);
  float* ecl = (float*)alloc((size_t)kECL * 16 * 4);
  float* aggcs = (float*)alloc((size_t)kNSite * kDCe * 4);
  float* hbuf  = (float*)alloc((size_t)kNSite * 80 * 4);
  float* pooled = (float*)alloc((size_t)kG * 80 * 4);
  float* cnt    = (float*)alloc((size_t)kG * 4);
  float* t1 = (float*)alloc((size_t)kG * 512 * 4);
  float* t2 = (float*)alloc((size_t)kG * 128 * 4);
  float* t3 = (float*)alloc((size_t)kG * 64 * 4);
  float* t4 = (float*)alloc((size_t)kG * 4);
  const size_t kSS = 5 * 5 * 512, kLC = 7 * 8 * 512, kGC = 6 * 5 * 512;
  half_t* pb[2][3][2];  // [layer][0=ss,1=lc,2=cl][0=Wf,1=Ws]
  for (int L = 0; L < 2; ++L) {
    pb[L][0][0] = (half_t*)alloc(kSS * 2); pb[L][0][1] = (half_t*)alloc(kSS * 2);
    pb[L][1][0] = (half_t*)alloc(kLC * 2); pb[L][1][1] = (half_t*)alloc(kLC * 2);
    pb[L][2][0] = (half_t*)alloc(kLC * 2); pb[L][2][1] = (half_t*)alloc(kLC * 2);
  }
  half_t* gcp = (half_t*)alloc(kGC * 2);

  const int B = 256;
  // ---- pack weights to f16 WMMA-B fragments ----
  for (int L = 0; L < 2; ++L) {
    pack_b<<<CDIV((int)kSS, B), B, 0, stream>>>(P(L*12+8),  pb[L][0][0], 160, 80, 5, 5);
    pack_b<<<CDIV((int)kSS, B), B, 0, stream>>>(P(L*12+9),  pb[L][0][1], 160, 80, 5, 5);
    pack_b<<<CDIV((int)kLC, B), B, 0, stream>>>(P(L*12+4),  pb[L][1][0], 240, 112, 8, 7);
    pack_b<<<CDIV((int)kLC, B), B, 0, stream>>>(P(L*12+5),  pb[L][1][1], 240, 112, 8, 7);
    pack_b<<<CDIV((int)kLC, B), B, 0, stream>>>(P(L*12+0),  pb[L][2][0], 240, 112, 8, 7);
    pack_b<<<CDIV((int)kLC, B), B, 0, stream>>>(P(L*12+1),  pb[L][2][1], 240, 112, 8, 7);
  }
  pack_b_gc<<<CDIV((int)kGC, B), B, 0, stream>>>(P(35), P(36), gcp);

  // ---- node features + edge RBFs ----
  build_site<<<CDIV(kNSite * kDSite, B), B, 0, stream>>>(site_el, site_ox, P(32), P(34), xsA);
  build_ce<<<CDIV(kNCe * kDCe, B), B, 0, stream>>>(ce_el, ce_ox, ce_geo, ce_csm,
                                                   P(32), P(34), P(33), P(39), P(40), xcA);
  build_lig<<<CDIV(kNLig * kDLig, B), B, 0, stream>>>(lig_el, lig_ox, lig_d, lig_a,
                                                      P(32), P(34), P(41), P(42), P(37), P(38), xlA);
  edge_rbf<<<CDIV(kELC * 16, B), B, 0, stream>>>(ea_lc, elc, kELC, P(43), P(44));
  edge_rbf<<<CDIV(kECL * 16, B), B, 0, stream>>>(ea_cl, ecl, kECL, P(43), P(44));

  // ---- 2 conv layers, ping-pong A<->B ----
  for (int L = 0; L < 2; ++L) {
    float *xs_in = L ? xsB : xsA, *xs_out = L ? xsA : xsB;
    float *xc_in = L ? xcB : xcA, *xc_out = L ? xcA : xcB;
    float *xl_in = L ? xlB : xlA, *xl_out = L ? xlA : xlB;
    hipMemsetAsync(xs_out, 0, (size_t)kNSite * kDSite * 4, stream);
    hipMemsetAsync(xc_out, 0, (size_t)kNCe * kDCe * 4, stream);
    hipMemsetAsync(xl_out, 0, (size_t)kNLig * kDLig * 4, stream);
    cgconv_edge<80, 80, 0, 5, 5><<<CDIV(CDIV(kESS, 16), 4), 128, 0, stream>>>(
        ei_ss, kESS, xs_in, xs_in, nullptr,
        pb[L][0][0], pb[L][0][1], P(L*12+10), P(L*12+11), xs_out);
    cgconv_edge<112, 112, 1, 8, 7><<<CDIV(CDIV(kELC, 16), 4), 128, 0, stream>>>(
        ei_lc, kELC, xc_in, xl_in, elc,
        pb[L][1][0], pb[L][1][1], P(L*12+6), P(L*12+7), xc_out);
    cgconv_edge<112, 112, 1, 8, 7><<<CDIV(CDIV(kECL, 16), 4), 128, 0, stream>>>(
        ei_cl, kECL, xl_in, xc_in, ecl,
        pb[L][2][0], pb[L][2][1], P(L*12+2), P(L*12+3), xl_out);
    residual_update<<<CDIV(kNSite * kDSite, B), B, 0, stream>>>(xs_in, xs_out, kNSite * kDSite, 0);
    residual_update<<<CDIV(kNCe * kDCe, B), B, 0, stream>>>(xc_in, xc_out, kNCe * kDCe, 1);
    residual_update<<<CDIV(kNLig * kDLig, B), B, 0, stream>>>(xl_in, xl_out, kNLig * kDLig, 1);
  }
  // final features now in the A buffers
  // ---- GraphConv ce->site ----
  hipMemsetAsync(aggcs, 0, (size_t)kNSite * kDCe * 4, stream);
  scatter_cs<<<CDIV(kECS * kDCe, B), B, 0, stream>>>(ei_cs, xcA, aggcs);
  gc_node_gemm<<<CDIV(CDIV(kNSite, 16), 4), 128, 0, stream>>>(aggcs, xsA, gcp, hbuf);

  // ---- global mean pool ----
  hipMemsetAsync(pooled, 0, (size_t)kG * 80 * 4, stream);
  hipMemsetAsync(cnt, 0, (size_t)kG * 4, stream);
  pool_sum<<<CDIV(kNSite * 80, B), B, 0, stream>>>(batch, hbuf, pooled);
  pool_cnt<<<CDIV(kNSite, B), B, 0, stream>>>(batch, cnt);
  pool_div<<<CDIV(kG * 80, B), B, 0, stream>>>(pooled, cnt);

  // ---- dense head + scaler ----
  dense_layer<<<CDIV(kG * 512, B), B, 0, stream>>>(pooled, P(24), P(25), t1, kG, 80, 512, 1);
  dense_layer<<<CDIV(kG * 128, B), B, 0, stream>>>(t1, P(26), P(27), t2, kG, 512, 128, 1);
  dense_layer<<<CDIV(kG * 64, B), B, 0, stream>>>(t2, P(28), P(29), t3, kG, 128, 64, 1);
  dense_layer<<<CDIV(kG, B), B, 0, stream>>>(t3, P(30), P(31), t4, kG, 64, 1, 0);
  scale_out<<<CDIV(kG, B), B, 0, stream>>>(t4, P(45), P(46), (float*)d_out, kG);
}
